// Model_29789893165726
// MI455X (gfx1250) — matrix operand order repaired and checked by reading, not compiled, not run on hardware
//
#include <hip/hip_runtime.h>

#define NN 100000
#define NE 200000
#define HH 128
#define NLVL 8
#define NGRP 21
#define TM 64

typedef __bf16 bf16;
typedef __attribute__((ext_vector_type(16))) __bf16 v16bf;
typedef __attribute__((ext_vector_type(8)))  __bf16 v8bf;
typedef __attribute__((ext_vector_type(8)))  float  v8f;

// ---------------------------------------------------------------------------
// helpers
// ---------------------------------------------------------------------------
static __device__ __forceinline__ v16bf cat8(v8bf lo, v8bf hi) {
  return __builtin_shufflevector(lo, hi, 0,1,2,3,4,5,6,7,8,9,10,11,12,13,14,15);
}

// A fragment (16x32 bf16, M=lane%16) from an LDS tile, row-major, stride ldk.
// ISA layout (16-bit A 16x32): lanes 0-15: K=k0+{0..7, 16..23}; lanes 16-31:
// K=k0+{8..15, 24..31}.  Each run of 8 bf16 is one 16B LDS load.
static __device__ __forceinline__ v16bf load_a(const bf16* base, int ldk, int lane, int k0) {
  int m  = lane & 15;
  int kb = (lane & 16) ? 8 : 0;
  const bf16* p = base + m * ldk + k0 + kb;
  v8bf lo = *(const v8bf*)(p);
  v8bf hi = *(const v8bf*)(p + 16);
  return cat8(lo, hi);
}

// B fragment from pre-swizzled global weights: per (tile, lane) the 16 bf16
// are contiguous (32B) -> two fully coalesced b128 loads, L2-resident.
static __device__ __forceinline__ v16bf load_b(const bf16* frag, int tile, int lane) {
  const bf16* p = frag + ((size_t)tile * 32 + lane) * 16;
  v8bf lo = *(const v8bf*)(p);
  v8bf hi = *(const v8bf*)(p + 8);
  return cat8(lo, hi);
}

#define WMMA_BF16(a, b, c) \
  __builtin_amdgcn_wmma_f32_16x16x32_bf16(false, (a), false, (b), (short)0, (c), false, false)

static __device__ __forceinline__ int gi_of_gate(int g) {
  return g == 3 ? 0 : g == 2 ? 1 : g == 5 ? 2 : -1;
}
static __device__ __forceinline__ int group_of(int gate, int lvl) {
  int gi = gi_of_gate(gate);
  return (gi >= 0 && lvl >= 1 && lvl < NLVL) ? (lvl - 1) * 3 + gi : -1;
}

// meta layout (ints): [0..20] node cnt/cursor, [32..53] node offsets(22),
//                     [64..84] edge cnt/cursor, [96..117] edge offsets(22)

// ---------------------------------------------------------------------------
// preprocessing kernels
// ---------------------------------------------------------------------------
__global__ void prep_meta_hstab(const float* Ws, const float* Wt, const float* hsW,
                                const float* hsb_, int* meta, float* hstab) {
  int t = threadIdx.x;                       // 256 threads, 1 block
  if (t < 128) meta[t] = 0;
  for (int o = t; o < 6 * HH; o += 256) {
    int g = o / HH, h = o % HH;
    float acc = hsb_[h];
    for (int k = 0; k < HH; ++k) acc += Ws[g * HH + k] * hsW[k * HH + h];
    for (int k = 0; k < HH; ++k) acc += Wt[g * HH + k] * hsW[(HH + k) * HH + h];
    hstab[o] = acc;
  }
}

__global__ void init_node_fields(const int* gate, const float* hstab, float* hs_out,
                                 float* hf_out, bf16* hsb, bf16* hfb, float* msg) {
  int idx = blockIdx.x * 256 + threadIdx.x;
  if (idx >= NN * HH) return;
  int i = idx >> 7, h = idx & 127;
  float v = hstab[gate[i] * HH + h];
  hs_out[idx] = v;
  hsb[idx]    = (bf16)v;
  hf_out[idx] = 0.f;
  hfb[idx]    = (bf16)0.f;
  msg[idx]    = 0.f;
}

__global__ void count_nodes(const int* gate, const int* lvl, int* meta) {
  int i = blockIdx.x * 256 + threadIdx.x;
  if (i >= NN) return;
  int g = group_of(gate[i], lvl[i]);
  if (g >= 0) atomicAdd(&meta[g], 1);
}

__global__ void count_edges(const int* gate, const int* lvl, const int* eidx, int* meta) {
  int e = blockIdx.x * 256 + threadIdx.x;
  if (e >= NE) return;
  int d = eidx[NE + e];
  int g = group_of(gate[d], lvl[d]);
  if (g >= 0) atomicAdd(&meta[64 + g], 1);
}

__global__ void scan_meta(int* meta) {
  if (threadIdx.x == 0 && blockIdx.x == 0) {
    int acc = 0;
    for (int g = 0; g < NGRP; ++g) { meta[32 + g] = acc; acc += meta[g]; meta[g] = 0; }
    meta[32 + NGRP] = acc;
    acc = 0;
    for (int g = 0; g < NGRP; ++g) { meta[96 + g] = acc; acc += meta[64 + g]; meta[64 + g] = 0; }
    meta[96 + NGRP] = acc;
  }
}

__global__ void place_nodes(const int* gate, const int* lvl, int* meta, int* nlist) {
  int i = blockIdx.x * 256 + threadIdx.x;
  if (i >= NN) return;
  int g = group_of(gate[i], lvl[i]);
  if (g < 0) return;
  int pos = atomicAdd(&meta[g], 1);
  nlist[meta[32 + g] + pos] = i;
}

__global__ void place_edges(const int* gate, const int* lvl, const int* eidx,
                            int* meta, int* elist) {
  int e = blockIdx.x * 256 + threadIdx.x;
  if (e >= NE) return;
  int d = eidx[NE + e];
  int g = group_of(gate[d], lvl[d]);
  if (g < 0) return;
  int pos = atomicAdd(&meta[64 + g], 1);
  elist[meta[96 + g] + pos] = e;
}

// Pre-swizzle a [K x Ncols] (or transposed-source) f32 weight matrix into the
// WMMA bf16 B-fragment order: tile=(kt*NT+nt), lane, h -> K=kt*32+(lane<16?0:8)+(h<8?h:h+8),
// N=nt*16+(lane&15).
__global__ void frag_weights(const float* src, bf16* dst, int K, int Ncols, int transposed) {
  int e = blockIdx.x * 256 + threadIdx.x;
  int total = K * Ncols;
  if (e >= total) return;
  int NT = Ncols >> 4;
  int tile = e >> 9;
  int r = e & 511;
  int lane = r >> 4, h = r & 15;
  int kt = tile / NT, nt = tile % NT;
  int k = kt * 32 + ((lane & 16) ? 8 : 0) + (h < 8 ? h : h + 8);
  int n = nt * 16 + (lane & 15);
  float v = transposed ? src[n * K + k] : src[k * Ncols + n];
  dst[e] = (bf16)v;
}

// ---------------------------------------------------------------------------
// edge MLP + scatter-add:  msg[dst] += W3*relu(W2*relu(W1*[hs,hf](src)+b1)+b2)+b3
// 64 edges per block; 8 waves; wave w owns Mtile=(w&3), N-half=(w>>2).
// ---------------------------------------------------------------------------
static __device__ __forceinline__ void mm_layer(const bf16* Atile, int lda, int ksteps,
                                                const bf16* wf, int NT, int lane,
                                                int nt0, v8f c[4]) {
  for (int kt = 0; kt < ksteps; ++kt) {
    v16bf a = load_a(Atile, lda, lane, kt * 32);
#pragma unroll
    for (int j = 0; j < 4; ++j) {
      v16bf b = load_b(wf, kt * NT + nt0 + j, lane);
      c[j] = WMMA_BF16(a, b, c[j]);
    }
  }
}

__global__ __launch_bounds__(256) void edge_mlp(
    int level, const int* meta, const int* elist, const int* eidx,
    const bf16* hsb, const bf16* hfb,
    const bf16* w1f, const bf16* w2f, const bf16* w3f,
    const float* b1a, const float* b2a, const float* b3a, float* msg) {
  int gi  = blockIdx.y;
  int grp = (level - 1) * 3 + gi;
  int ebase = meta[96 + grp];
  int cnt   = meta[96 + grp + 1] - ebase;
  int row0  = blockIdx.x * TM;
  if (row0 >= cnt) return;

  const bf16* w1 = w1f + (size_t)gi * 256 * 128;
  const bf16* w2 = w2f + (size_t)gi * 128 * 128;
  const bf16* w3 = w3f + (size_t)gi * 128 * 128;
  const float* b1 = b1a + gi * HH;
  const float* b2 = b2a + gi * HH;
  const float* b3 = b3a + gi * HH;

  __shared__ __align__(16) bf16 Xs[64][264];   // X, then reused as Y2
  __shared__ __align__(16) bf16 Ys[64][136];   // Y1
  __shared__ int dsts[64];

  int t = threadIdx.x;
  {
    // gather 64 rows of [hs,hf](src) as bf16; 4 threads per row
    int r = t >> 2, q = t & 3;
    int gl = row0 + r;
    int s = -1, d = -1;
    if (gl < cnt) { int e = elist[ebase + gl]; s = eidx[e]; d = eidx[NE + e]; }
    if (q == 0) dsts[r] = d;
    const bf16* srow = nullptr;
    if (s >= 0) srow = (q < 2) ? hsb + (size_t)s * HH + q * 64
                               : hfb + (size_t)s * HH + (q - 2) * 64;
#pragma unroll
    for (int i = 0; i < 8; ++i) {
      v8bf v = {};
      if (srow) v = *(const v8bf*)(srow + i * 8);
      *(v8bf*)(&Xs[r][q * 64 + i * 8]) = v;
    }
  }
  __syncthreads();

  int lane = t & 31, wav = t >> 5;
  int mt  = wav & 3;
  int nt0 = (wav >> 2) * 4;

  // ----- layer 1: [64,256] x [256,128] -----
  v8f c[4] = {};
  mm_layer(&Xs[mt * 16][0], 264, 8, w1, 8, lane, nt0, c);
  {
    int mb = (lane & 16) ? 8 : 0, nl = lane & 15;
#pragma unroll
    for (int j = 0; j < 4; ++j) {
      int n = (nt0 + j) * 16 + nl;
      float bias = b1[n];
#pragma unroll
      for (int v = 0; v < 8; ++v) {
        float val = c[j][v] + bias;
        Ys[mt * 16 + mb + v][n] = (bf16)(val > 0.f ? val : 0.f);
      }
    }
  }
  __syncthreads();

  // ----- layer 2: [64,128] x [128,128] -----
  v8f c2[4] = {};
  mm_layer(&Ys[mt * 16][0], 136, 4, w2, 8, lane, nt0, c2);
  {
    int mb = (lane & 16) ? 8 : 0, nl = lane & 15;
#pragma unroll
    for (int j = 0; j < 4; ++j) {
      int n = (nt0 + j) * 16 + nl;
      float bias = b2[n];
#pragma unroll
      for (int v = 0; v < 8; ++v) {
        float val = c2[j][v] + bias;
        Xs[mt * 16 + mb + v][n] = (bf16)(val > 0.f ? val : 0.f);  // Y2 in Xs
      }
    }
  }
  __syncthreads();

  // ----- layer 3 + scatter-add -----
  v8f c3[4] = {};
  mm_layer(&Xs[mt * 16][0], 264, 4, w3, 8, lane, nt0, c3);
  {
    int mb = (lane & 16) ? 8 : 0, nl = lane & 15;
#pragma unroll
    for (int j = 0; j < 4; ++j) {
      int n = (nt0 + j) * 16 + nl;
      float bias = b3[n];
#pragma unroll
      for (int v = 0; v < 8; ++v) {
        int lrow = mt * 16 + mb + v;
        if (row0 + lrow < cnt) {
          int d = dsts[lrow];
          atomicAdd(&msg[(size_t)d * HH + n], c3[j][v] + bias);
        }
      }
    }
  }
}

// ---------------------------------------------------------------------------
// GRU update (h_old == 0 for every node at its own level):
//   gin = msg @ wih.T + bih ;  gh == bhh
//   hf = (1-sigmoid(i_z+bhh_z)) * tanh(i_n + sigmoid(i_r+bhh_r)*bhh_n)
// 16 nodes per block; 4 waves; wave w owns ntiles w*6..w*6+5 of 24 (N=384).
// ---------------------------------------------------------------------------
__global__ __launch_bounds__(128) void gru_upd(
    int level, const int* meta, const int* nlist, const float* msg,
    const bf16* wihf, const float* biha, const float* bhha,
    float* hf, bf16* hfb) {
  int gi  = blockIdx.y;
  int grp = (level - 1) * 3 + gi;
  int nbase = meta[32 + grp];
  int cnt   = meta[32 + grp + 1] - nbase;
  int m0    = blockIdx.x * 16;
  if (m0 >= cnt) return;

  const bf16* wih = wihf + (size_t)gi * 128 * 384;
  const float* bih = biha + gi * 3 * HH;
  const float* bhh = bhha + gi * 3 * HH;

  __shared__ __align__(16) bf16 As[16][136];
  __shared__ float gin[16][392];
  __shared__ int nodes[16];

  int t = threadIdx.x;
  {
    int r = t >> 3, p = t & 7;   // 8 threads per row, 16 cols each
    int gl = m0 + r;
    int node = (gl < cnt) ? nlist[nbase + gl] : -1;
    if (p == 0) nodes[r] = node;
#pragma unroll
    for (int i = 0; i < 16; ++i) {
      float v = (node >= 0) ? msg[(size_t)node * HH + p * 16 + i] : 0.f;
      As[r][p * 16 + i] = (bf16)v;
    }
  }
  __syncthreads();

  int lane = t & 31, wav = t >> 5;
  v8f c[6] = {};
  for (int kt = 0; kt < 4; ++kt) {
    v16bf a = load_a(&As[0][0], 136, lane, kt * 32);
#pragma unroll
    for (int j = 0; j < 6; ++j) {
      v16bf b = load_b(wih, kt * 24 + wav * 6 + j, lane);
      c[j] = WMMA_BF16(a, b, c[j]);
    }
  }
  {
    int mb = (lane & 16) ? 8 : 0, nl = lane & 15;
#pragma unroll
    for (int j = 0; j < 6; ++j) {
      int n = (wav * 6 + j) * 16 + nl;
      float bias = bih[n];
#pragma unroll
      for (int v = 0; v < 8; ++v) gin[mb + v][n] = c[j][v] + bias;
    }
  }
  __syncthreads();

  {
    int h = t;  // 0..127: one output column per thread, all 16 rows
    float hr = bhh[h], hz = bhh[HH + h], hn = bhh[2 * HH + h];
    for (int m = 0; m < 16; ++m) {
      if (m0 + m >= cnt) break;
      float ir = gin[m][h], iz = gin[m][HH + h], inn = gin[m][2 * HH + h];
      float r = 1.f / (1.f + __expf(-(ir + hr)));
      float z = 1.f / (1.f + __expf(-(iz + hz)));
      float nst = tanhf(inn + r * hn);
      float out = (1.f - z) * nst;  // + z*h_old, h_old==0
      int node = nodes[m];
      hf[(size_t)node * HH + h]  = out;
      hfb[(size_t)node * HH + h] = (bf16)out;
    }
  }
}

// ---------------------------------------------------------------------------
// host launcher
// ---------------------------------------------------------------------------
extern "C" void kernel_launch(void* const* d_in, const int* in_sizes, int n_in,
                              void* d_out, int out_size, void* d_ws, size_t ws_size,
                              hipStream_t stream) {
  (void)in_sizes; (void)n_in; (void)out_size; (void)ws_size;
  const int*   gate   = (const int*)d_in[0];
  const int*   lvl    = (const int*)d_in[1];
  const int*   eidx   = (const int*)d_in[2];
  const float* Ws     = (const float*)d_in[3];
  const float* Wt     = (const float*)d_in[4];
  const float* hsW    = (const float*)d_in[5];
  const float* hs_b   = (const float*)d_in[6];
  const float* mlp_w1 = (const float*)d_in[7];
  const float* mlp_b1 = (const float*)d_in[8];
  const float* mlp_w2 = (const float*)d_in[9];
  const float* mlp_b2 = (const float*)d_in[10];
  const float* mlp_w3 = (const float*)d_in[11];
  const float* mlp_b3 = (const float*)d_in[12];
  const float* gru_wih = (const float*)d_in[13];
  const float* gru_bih = (const float*)d_in[15];
  const float* gru_bhh = (const float*)d_in[16];

  char* w = (char*)d_ws;
  size_t off = 0;
  auto take = [&](size_t bytes) -> char* {
    char* p = w + off;
    off = (off + bytes + 255) & ~(size_t)255;
    return p;
  };
  float* msg   = (float*)take((size_t)NN * HH * 4);
  bf16*  hsb   = (bf16*) take((size_t)NN * HH * 2);
  bf16*  hfb   = (bf16*) take((size_t)NN * HH * 2);
  float* hstab = (float*)take(6 * HH * 4);
  int*   meta  = (int*)  take(128 * 4);
  int*   nlist = (int*)  take((size_t)NN * 4);
  int*   elist = (int*)  take((size_t)NE * 4);
  bf16*  w1f   = (bf16*) take((size_t)3 * 256 * 128 * 2);
  bf16*  w2f   = (bf16*) take((size_t)3 * 128 * 128 * 2);
  bf16*  w3f   = (bf16*) take((size_t)3 * 128 * 128 * 2);
  bf16*  wihf  = (bf16*) take((size_t)3 * 128 * 384 * 2);

  float* hs_out = (float*)d_out;
  float* hf_out = hs_out + (size_t)NN * HH;

  // ---- preprocessing ----
  prep_meta_hstab<<<1, 256, 0, stream>>>(Ws, Wt, hsW, hs_b, meta, hstab);
  init_node_fields<<<(NN * HH + 255) / 256, 256, 0, stream>>>(gate, hstab, hs_out,
                                                              hf_out, hsb, hfb, msg);
  count_nodes<<<(NN + 255) / 256, 256, 0, stream>>>(gate, lvl, meta);
  count_edges<<<(NE + 255) / 256, 256, 0, stream>>>(gate, lvl, eidx, meta);
  scan_meta<<<1, 64, 0, stream>>>(meta);
  place_nodes<<<(NN + 255) / 256, 256, 0, stream>>>(gate, lvl, meta, nlist);
  place_edges<<<(NE + 255) / 256, 256, 0, stream>>>(gate, lvl, eidx, meta, elist);

  for (int gi = 0; gi < 3; ++gi) {
    frag_weights<<<(256 * 128 + 255) / 256, 256, 0, stream>>>(
        mlp_w1 + (size_t)gi * 256 * 128, w1f + (size_t)gi * 256 * 128, 256, 128, 0);
    frag_weights<<<(128 * 128 + 255) / 256, 256, 0, stream>>>(
        mlp_w2 + (size_t)gi * 128 * 128, w2f + (size_t)gi * 128 * 128, 128, 128, 0);
    frag_weights<<<(128 * 128 + 255) / 256, 256, 0, stream>>>(
        mlp_w3 + (size_t)gi * 128 * 128, w3f + (size_t)gi * 128 * 128, 128, 128, 0);
    frag_weights<<<(128 * 384 + 255) / 256, 256, 0, stream>>>(
        gru_wih + (size_t)gi * 384 * 128, wihf + (size_t)gi * 128 * 384, 128, 384, 1);
  }

  // ---- level-synchronous propagation ----
  dim3 eg((NE + TM - 1) / TM, 3);
  dim3 gg((NN + 15) / 16, 3);
  for (int level = 1; level < NLVL; ++level) {
    edge_mlp<<<eg, 256, 0, stream>>>(level, meta, elist, eidx, hsb, hfb,
                                     w1f, w2f, w3f, mlp_b1, mlp_b2, mlp_b3, msg);
    gru_upd<<<gg, 128, 0, stream>>>(level, meta, nlist, msg, wihf,
                                    gru_bih, gru_bhh, hf_out, hfb);
  }
}
